// ALMSLayer_91104846283496
// MI455X (gfx1250) — matrix operand order repaired and checked by reading, not compiled, not run on hardware
//
#include <hip/hip_runtime.h>
#include <hip/hip_bf16.h>

// ALMS layer for MI455X (gfx1250, wave32, WMMA).
// B=8192 nodes, D=128 features, K=32 neighbors.
// All three B*B*D GEMMs run on v_wmma_f32_16x16x32_bf16; the softmax GEMM is
// fused flash-attention style (column-split x4 + merge for occupancy) so the
// 8192x8192 score matrix never touches HBM. Value tiles are staged with
// global_load_async_to_lds_b128 (ASYNCcnt path); B-fragment loads are grouped
// ahead of each WMMA burst with __builtin_amdgcn_sched_barrier so the burst
// issues back-to-back instead of load/wait/wmma lock-step.

#define B_N   8192
#define D_DIM 128
#define KNN   32
#define LAMBDA_REG 0.1f
#define INV_TEMP   10.0f   // 1/TEMP
#define NSPLIT 4
#define CSPAN  (B_N / NSPLIT)   // 2048 columns per flash split

typedef __attribute__((ext_vector_type(16))) __bf16 v16bf;
typedef __attribute__((ext_vector_type(8)))  float  v8f;

union FragAB { v16bf v; uint4 q[2]; };

// round-to-nearest-even float -> bf16 bits (avoid relying on __bf16 scalar ops)
__device__ __forceinline__ unsigned short f2bf(float f) {
    unsigned u = __float_as_uint(f);
    u += 0x7FFFu + ((u >> 16) & 1u);
    return (unsigned short)(u >> 16);
}

__device__ __forceinline__ float blockReduceSum128(float v, float* red4) {
    #pragma unroll
    for (int m = 16; m >= 1; m >>= 1) v += __shfl_xor(v, m, 32);
    int wid = threadIdx.x >> 5;
    if ((threadIdx.x & 31) == 0) red4[wid] = v;
    __syncthreads();
    float s = red4[0] + red4[1] + red4[2] + red4[3];
    __syncthreads();
    return s;
}

__device__ __forceinline__ void loadB(FragAB& b, const unsigned short* p) {
    b.q[0] = *(const uint4*)(p);
    b.q[1] = *(const uint4*)(p + 8);
}

// ---------------------------------------------------------------------------
// Kernel 1: row L2 norms; emit bf16 x_hat = f/max(||f||,1e-12) and bf16 f.
// ---------------------------------------------------------------------------
__global__ __launch_bounds__(128) void k_normalize(const float* __restrict__ f,
                                                   unsigned short* __restrict__ xb,
                                                   unsigned short* __restrict__ fb,
                                                   float* __restrict__ fnorm) {
    __shared__ float red4[4];
    int row = blockIdx.x, d = threadIdx.x;
    float v = f[(size_t)row * D_DIM + d];
    float n = sqrtf(blockReduceSum128(v * v, red4));
    float inv = 1.0f / fmaxf(n, 1e-12f);
    xb[(size_t)row * D_DIM + d] = f2bf(v * inv);
    fb[(size_t)row * D_DIM + d] = f2bf(v);
    if (d == 0) fnorm[row] = n;
}

// ---------------------------------------------------------------------------
// Kernel 2: sim = x_hat x_hat^T via WMMA, streaming top-32 per row (self
// excluded; equivalent to top-33 then dropping self since self-sim = 1.0 max).
// Block = 64 thr (2 waves), 32 rows/block -> 256 blocks; 32-column tiles.
// ---------------------------------------------------------------------------
__global__ __launch_bounds__(64, 1) void k_topk(const unsigned short* __restrict__ xb,
                                                int* __restrict__ knn_idx) {
    __shared__ __attribute__((aligned(16))) unsigned short Xt[D_DIM][32]; // 8KB, Xt[k][c]=x[col0+c][k]
    __shared__ float Sbuf[2][16][32];                                     // 4KB
    __shared__ float topv[32][KNN];                                       // 4KB
    __shared__ int   topi[32][KNN];                                       // 4KB
    int tid = threadIdx.x, wave = tid >> 5, lane = tid & 31;
    int waveRow0 = blockIdx.x * 32 + wave * 16;

    for (int i = tid; i < 32 * KNN; i += 64) { (&topv[0][0])[i] = -3.0e38f; (&topi[0][0])[i] = 0; }

    float minv = -3.0e38f; int minp = 0;

    // Resident A fragments: this wave's 16 rows, K=128 -> 4 fragments.
    // A layout: lane m=l&15; elems 0-7: K = hs*8+0..7, elems 8-15: K = 16+hs*8+0..7.
    FragAB afrag[4];
    {
        int m = lane & 15, hs = (lane >> 4) & 1;
        const unsigned short* rp = xb + (size_t)(waveRow0 + m) * D_DIM;
        #pragma unroll
        for (int kt = 0; kt < 4; ++kt) {
            int kb = kt * 32 + hs * 8;
            afrag[kt].q[0] = *(const uint4*)(rp + kb);
            afrag[kt].q[1] = *(const uint4*)(rp + kb + 16);
        }
    }

    for (int col0 = 0; col0 < B_N; col0 += 32) {
        // prefetch next tile into near caches (L2-resident working set)
        if (col0 + 32 < B_N) {
            int r = tid & 31, half = (tid >> 5) & 1;
            __builtin_prefetch(xb + (size_t)(col0 + 32 + r) * D_DIM + half * 64, 0, 0);
        }
        __syncthreads();
        // Stage transposed tile: Xt[k][c] = xb[col0+c][k] (vector global loads)
        #pragma unroll
        for (int i = 0; i < 8; ++i) {
            int q = tid + i * 64;           // 512 quads of 8 bf16
            int c = q >> 4, qc = q & 15;
            uint4 d4 = *(const uint4*)(xb + (size_t)(col0 + c) * D_DIM + qc * 8);
            const unsigned short* u = (const unsigned short*)&d4;
            #pragma unroll
            for (int j = 0; j < 8; ++j) Xt[qc * 8 + j][c] = u[j];
        }
        __syncthreads();
        // S tile 16x32: preload all 8 B fragments (pinned by sched_barrier),
        // then 8 back-to-back WMMAs.
        {
            FragAB bfr[8];
            #pragma unroll
            for (int kt = 0; kt < 4; ++kt) {
                const unsigned short* bp = xb + (size_t)(col0 + (lane & 15)) * D_DIM + kt * 32 + ((lane >> 4) & 1) * 8;
                bfr[2 * kt].q[0]     = *(const uint4*)(bp);
                bfr[2 * kt].q[1]     = *(const uint4*)(bp + 16);
                bfr[2 * kt + 1].q[0] = *(const uint4*)(bp + 16 * D_DIM);
                bfr[2 * kt + 1].q[1] = *(const uint4*)(bp + 16 * D_DIM + 16);
            }
            __builtin_amdgcn_sched_barrier(0);
            v8f c0v = {}, c1v = {};
            #pragma unroll
            for (int kt = 0; kt < 4; ++kt) {
                c0v = __builtin_amdgcn_wmma_f32_16x16x32_bf16(false, afrag[kt].v, false, bfr[2 * kt].v,
                                                              (short)0, c0v, false, false);
                c1v = __builtin_amdgcn_wmma_f32_16x16x32_bf16(false, afrag[kt].v, false, bfr[2 * kt + 1].v,
                                                              (short)0, c1v, false, false);
            }
            __builtin_amdgcn_sched_barrier(0);
            int n = lane & 15, mo = (lane >> 4) * 8;
            #pragma unroll
            for (int r = 0; r < 8; ++r) {
                Sbuf[wave][mo + r][n]      = c0v[r];
                Sbuf[wave][mo + r][16 + n] = c1v[r];
            }
        }
        __syncthreads();
        // Streaming top-32 insertion: lanes 0..15 own one row each.
        if (lane < 16) {
            int row = waveRow0 + lane;
            float* tv = topv[wave * 16 + lane];
            int*   ti = topi[wave * 16 + lane];
            for (int cc = 0; cc < 32; ++cc) {
                int col = col0 + cc;
                if (col == row) continue;
                float v = Sbuf[wave][lane][cc];
                if (v > minv) {
                    tv[minp] = v; ti[minp] = col;
                    float mv = tv[0]; int mp = 0;
                    #pragma unroll
                    for (int t = 1; t < KNN; ++t) { if (tv[t] < mv) { mv = tv[t]; mp = t; } }
                    minv = mv; minp = mp;
                }
            }
        }
    }
    __syncthreads();
    if (lane < 16) {
        int row = waveRow0 + lane;
        for (int t = 0; t < KNN; ++t) knn_idx[(size_t)row * KNN + t] = topi[wave * 16 + lane][t];
    }
}

// ---------------------------------------------------------------------------
// Kernel 3: out += A_sym @ vin, A_sym = (E + E^T)/32, edge weight exactly 1/32
// (deg == K for every node in the reference). out must be pre-zeroed.
// ---------------------------------------------------------------------------
__global__ __launch_bounds__(128) void k_spmm(const int* __restrict__ knn,
                                              const float* __restrict__ vin,
                                              float* __restrict__ vout) {
    int i = blockIdx.x, d = threadIdx.x;
    const float w = 1.0f / (float)KNN;
    float self = vin[(size_t)i * D_DIM + d] * w;
    float acc = 0.0f;
    for (int t = 0; t < KNN; ++t) {
        int j = knn[(size_t)i * KNN + t];
        acc += vin[(size_t)j * D_DIM + d];                 // gather: edges (i -> j)
        atomicAdd(&vout[(size_t)j * D_DIM + d], self);     // scatter: edges (j <- i)
    }
    atomicAdd(&vout[(size_t)i * D_DIM + d], acc * w);
}

// ---------------------------------------------------------------------------
// Kernel 4: y_j = x_hat_j + 0.1 * g_j/max(||g_j||, eps)  (bf16)
// (raw[i,j] = sim + 0.1*cos = x_hat_i . y_j)
// ---------------------------------------------------------------------------
__global__ __launch_bounds__(128) void k_prep_y(const float* __restrict__ f,
                                                const float* __restrict__ g,
                                                const float* __restrict__ fnorm,
                                                unsigned short* __restrict__ yb) {
    __shared__ float red4[4];
    int row = blockIdx.x, d = threadIdx.x;
    float gv = g[(size_t)row * D_DIM + d];
    float gn = sqrtf(blockReduceSum128(gv * gv, red4));
    float fv = f[(size_t)row * D_DIM + d];
    float xh = fv / fmaxf(fnorm[row], 1e-12f);
    float gh = gv / fmaxf(gn, 1e-8f);
    yb[(size_t)row * D_DIM + d] = f2bf(xh + LAMBDA_REG * gh);
}

// ---------------------------------------------------------------------------
// Kernel 5: flash-softmax GEMM, column-split x NSPLIT for occupancy.
// Partial over columns [split*CSPAN, (split+1)*CSPAN):
//   Opart = sum_j exp(s_j - m) f_j  (unnormalized), plus per-row (m, l).
// Block = 4 waves * 16 rows; 32-col tiles; 16 WMMAs/tile/wave.
// ---------------------------------------------------------------------------
__global__ __launch_bounds__(128, 1) void k_flash(const unsigned short* __restrict__ xb,
                                                  const unsigned short* __restrict__ yb,
                                                  const unsigned short* __restrict__ fb,
                                                  float* __restrict__ Opart,
                                                  float* __restrict__ mlpart) {
    __shared__ __attribute__((aligned(16))) unsigned short Yt[D_DIM][32];   // 8KB transposed score tile
    __shared__ __attribute__((aligned(16))) unsigned short Fv[32][D_DIM];   // 8KB value tile (natural)
    __shared__ __attribute__((aligned(16))) unsigned short Pb[4][16][32];   // 4KB P per wave
    int tid = threadIdx.x, wave = tid >> 5, lane = tid & 31;
    int split = blockIdx.x >> 7;                  // 0..NSPLIT-1
    int waveRow0 = (blockIdx.x & 127) * 64 + wave * 16;
    int cbase = split * CSPAN;
    int m = lane & 15, hs = (lane >> 4) & 1;
    // LDS byte offset of Fv (generic shared pointer: low 32 bits are LDS addr)
    unsigned fvbase = (unsigned)(uintptr_t)(&Fv[0][0]);

    FragAB afrag[4];
    {
        const unsigned short* rp = xb + (size_t)(waveRow0 + m) * D_DIM;
        #pragma unroll
        for (int kt = 0; kt < 4; ++kt) {
            int kb = kt * 32 + hs * 8;
            afrag[kt].q[0] = *(const uint4*)(rp + kb);
            afrag[kt].q[1] = *(const uint4*)(rp + kb + 16);
        }
    }

    v8f O[8];
    #pragma unroll
    for (int j = 0; j < 8; ++j) { v8f z = {}; O[j] = z; }
    float mrow[8], lrow[8];
    #pragma unroll
    for (int r = 0; r < 8; ++r) { mrow[r] = -3.0e38f; lrow[r] = 0.0f; }

    for (int col0 = cbase; col0 < cbase + CSPAN; col0 += 32) {
        // prefetch next tile (y and f rows) into near caches
        if (col0 + 32 < cbase + CSPAN) {
            int r = tid & 31, half = (tid >> 5) & 1;
            const unsigned short* p = ((tid >> 6) ? fb : yb)
                                      + (size_t)(col0 + 32 + r) * D_DIM + half * 64;
            __builtin_prefetch(p, 0, 0);
        }
        __syncthreads();
        // --- value tile: async global -> LDS (ASYNCcnt path), 16B per lane ---
        {
            #pragma unroll
            for (int i = 0; i < 4; ++i) {
                int chunk = (wave * 4 + i) * 32 + lane;                 // 512 x 16B = 8KB
                unsigned lds_addr = fvbase + (unsigned)chunk * 16u;
                unsigned long long gaddr =
                    (unsigned long long)(uintptr_t)(fb + (size_t)col0 * D_DIM + (size_t)chunk * 8);
                asm volatile("global_load_async_to_lds_b128 %0, %1, off"
                             :: "v"(lds_addr), "v"(gaddr) : "memory");
            }
        }
        // --- score tile: manual transposed stage Yt[k][c] = yb[col0+c][k] ---
        #pragma unroll
        for (int i = 0; i < 4; ++i) {
            int q = tid + i * 128;
            int c = q >> 4, qc = q & 15;
            uint4 d4 = *(const uint4*)(yb + (size_t)(col0 + c) * D_DIM + qc * 8);
            const unsigned short* u = (const unsigned short*)&d4;
            #pragma unroll
            for (int j = 0; j < 8; ++j) Yt[qc * 8 + j][c] = u[j];
        }
        asm volatile("s_wait_asynccnt 0x0" ::: "memory");
        __syncthreads();

        // scores: two 16x16 tiles, K=128; two 4-fragment load batches,
        // each pinned ahead of its WMMA burst.
        v8f s0 = {}, s1 = {};
        #pragma unroll
        for (int kc = 0; kc < 2; ++kc) {
            FragAB bfr[4];
            const unsigned short* yp = yb + (size_t)(col0 + m) * D_DIM + (2 * kc) * 32 + hs * 8;
            bfr[0].q[0] = *(const uint4*)(yp);
            bfr[0].q[1] = *(const uint4*)(yp + 16);
            bfr[1].q[0] = *(const uint4*)(yp + 16 * D_DIM);
            bfr[1].q[1] = *(const uint4*)(yp + 16 * D_DIM + 16);
            bfr[2].q[0] = *(const uint4*)(yp + 32);
            bfr[2].q[1] = *(const uint4*)(yp + 48);
            bfr[3].q[0] = *(const uint4*)(yp + 16 * D_DIM + 32);
            bfr[3].q[1] = *(const uint4*)(yp + 16 * D_DIM + 48);
            __builtin_amdgcn_sched_barrier(0);
            s0 = __builtin_amdgcn_wmma_f32_16x16x32_bf16(false, afrag[2 * kc].v,     false, bfr[0].v,
                                                         (short)0, s0, false, false);
            s1 = __builtin_amdgcn_wmma_f32_16x16x32_bf16(false, afrag[2 * kc].v,     false, bfr[1].v,
                                                         (short)0, s1, false, false);
            s0 = __builtin_amdgcn_wmma_f32_16x16x32_bf16(false, afrag[2 * kc + 1].v, false, bfr[2].v,
                                                         (short)0, s0, false, false);
            s1 = __builtin_amdgcn_wmma_f32_16x16x32_bf16(false, afrag[2 * kc + 1].v, false, bfr[3].v,
                                                         (short)0, s1, false, false);
            __builtin_amdgcn_sched_barrier(0);
        }

        // online softmax; C-layout: VGPR r <-> row (mo+r), lanes <-> columns,
        // so xor-shuffles with masks 8..1 reduce within each 16-lane half.
        float scale[8];
        #pragma unroll
        for (int r = 0; r < 8; ++r) {
            float a = s0[r] * INV_TEMP, b = s1[r] * INV_TEMP;
            float mx = fmaxf(a, b);
            #pragma unroll
            for (int msk = 8; msk >= 1; msk >>= 1) mx = fmaxf(mx, __shfl_xor(mx, msk, 32));
            float mnew = fmaxf(mrow[r], mx);
            scale[r] = __expf(mrow[r] - mnew);
            mrow[r] = mnew;
            float p0 = __expf(a - mnew), p1 = __expf(b - mnew);
            s0[r] = p0; s1[r] = p1;
            float rs = p0 + p1;
            #pragma unroll
            for (int msk = 8; msk >= 1; msk >>= 1) rs += __shfl_xor(rs, msk, 32);
            lrow[r] = lrow[r] * scale[r] + rs;
        }
        #pragma unroll
        for (int j = 0; j < 8; ++j)
            #pragma unroll
            for (int r = 0; r < 8; ++r) O[j][r] *= scale[r];

        // P (16x32) to LDS as bf16, re-read as A fragment
        {
            int n = lane & 15, mo = (lane >> 4) * 8;
            #pragma unroll
            for (int r = 0; r < 8; ++r) {
                Pb[wave][mo + r][n]      = f2bf(s0[r]);
                Pb[wave][mo + r][16 + n] = f2bf(s1[r]);
            }
        }
        __syncthreads();
        FragAB pf;
        {
            const unsigned short* pp = &Pb[wave][m][hs * 8];
            pf.q[0] = *(const uint4*)(pp);
            pf.q[1] = *(const uint4*)(pp + 16);
        }
        // O += P @ V: V fragments gathered per column from natural-layout Fv in table k order;
        // two 4-fragment load batches pinned ahead of WMMA bursts.
        #pragma unroll
        for (int jc = 0; jc < 2; ++jc) {
            FragAB vfr[4];
            #pragma unroll
            for (int t = 0; t < 4; ++t)
                #pragma unroll
                for (int s = 0; s < 16; ++s)
                    ((unsigned short*)&vfr[t])[s] = Fv[(s < 8 ? hs * 8 + s : 16 + hs * 8 + (s - 8))][(jc * 4 + t) * 16 + m];
            __builtin_amdgcn_sched_barrier(0);
            #pragma unroll
            for (int t = 0; t < 4; ++t)
                O[jc * 4 + t] = __builtin_amdgcn_wmma_f32_16x16x32_bf16(false, pf.v, false, vfr[t].v,
                                                                        (short)0, O[jc * 4 + t], false, false);
            __builtin_amdgcn_sched_barrier(0);
        }
    }

    // epilogue: store unnormalized partial O and per-row (m, l)
    {
        int n = lane & 15, mo = (lane >> 4) * 8;
        #pragma unroll
        for (int r = 0; r < 8; ++r) {
            int row = waveRow0 + mo + r;
            size_t rowoff = ((size_t)split * B_N + row) * D_DIM;
            #pragma unroll
            for (int j = 0; j < 8; ++j) Opart[rowoff + j * 16 + n] = O[j][r];
            if (n == 0) {
                mlpart[((size_t)split * B_N + row) * 2 + 0] = mrow[r];
                mlpart[((size_t)split * B_N + row) * 2 + 1] = lrow[r];
            }
        }
    }
}

// ---------------------------------------------------------------------------
// Kernel 6: merge NSPLIT flash partials.
// out[row] = sum_s exp(m_s - M) * O_s[row] / sum_s exp(m_s - M) * l_s
// ---------------------------------------------------------------------------
__global__ __launch_bounds__(128) void k_merge(const float* __restrict__ Opart,
                                               const float* __restrict__ mlpart,
                                               float* __restrict__ out) {
    int row = blockIdx.x, d = threadIdx.x;
    float ms[NSPLIT], ls[NSPLIT], M = -3.0e38f;
    #pragma unroll
    for (int s = 0; s < NSPLIT; ++s) {
        ms[s] = mlpart[((size_t)s * B_N + row) * 2 + 0];
        ls[s] = mlpart[((size_t)s * B_N + row) * 2 + 1];
        M = fmaxf(M, ms[s]);
    }
    float L = 0.0f, acc = 0.0f;
    #pragma unroll
    for (int s = 0; s < NSPLIT; ++s) {
        float w = __expf(ms[s] - M);
        L   += ls[s] * w;
        acc += Opart[((size_t)s * B_N + row) * D_DIM + d] * w;
    }
    out[(size_t)row * D_DIM + d] = acc / L;
}

// ---------------------------------------------------------------------------
extern "C" void kernel_launch(void* const* d_in, const int* in_sizes, int n_in,
                              void* d_out, int out_size, void* d_ws, size_t ws_size,
                              hipStream_t stream) {
    (void)in_sizes; (void)n_in; (void)out_size; (void)ws_size;
    const float* features = (const float*)d_in[0];
    float* out = (float*)d_out;

    // workspace carve-out (~31.5 MB total), 256B aligned slices
    char* ws = (char*)d_ws;
    size_t off = 0;
    auto carve = [&](size_t bytes) { void* p = ws + off; off += (bytes + 255) & ~(size_t)255; return p; };
    unsigned short* xb   = (unsigned short*)carve((size_t)B_N * D_DIM * 2);   // bf16 x_hat
    unsigned short* fb   = (unsigned short*)carve((size_t)B_N * D_DIM * 2);   // bf16 features
    unsigned short* yb   = (unsigned short*)carve((size_t)B_N * D_DIM * 2);   // bf16 y
    float*          fnrm = (float*)carve((size_t)B_N * 4);
    int*            knn  = (int*)carve((size_t)B_N * KNN * 4);
    float*          dif1 = (float*)carve((size_t)B_N * D_DIM * 4);
    float*          geod = (float*)carve((size_t)B_N * D_DIM * 4);
    float*          Opar = (float*)carve((size_t)NSPLIT * B_N * D_DIM * 4);   // 16 MB
    float*          mlp  = (float*)carve((size_t)NSPLIT * B_N * 2 * 4);

    k_normalize<<<B_N, 128, 0, stream>>>(features, xb, fb, fnrm);
    k_topk<<<B_N / 32, 64, 0, stream>>>(xb, knn);
    hipMemsetAsync(dif1, 0, (size_t)B_N * D_DIM * 4, stream);
    k_spmm<<<B_N, 128, 0, stream>>>(knn, features, dif1);
    hipMemsetAsync(geod, 0, (size_t)B_N * D_DIM * 4, stream);
    k_spmm<<<B_N, 128, 0, stream>>>(knn, dif1, geod);
    k_prep_y<<<B_N, 128, 0, stream>>>(features, geod, fnrm, yb);
    k_flash<<<(B_N / 64) * NSPLIT, 128, 0, stream>>>(xb, yb, fb, Opar, mlp);
    k_merge<<<B_N, 128, 0, stream>>>(Opar, mlp, out);
}
